// MorletWaveletKanLayer_67465346285663
// MI455X (gfx1250) — matrix operand order repaired and checked by reading, not compiled, not run on hardware
//
#include <hip/hip_runtime.h>
#include <hip/hip_bf16.h>

// MorletWaveletKAN layer for MI455X (gfx1250, wave32).
// B=4096, I=128, O=512 derived at launch from in_sizes.
//
// Kernel 1: psi = cos(5u)*exp(-u^2/2), u = (x-bias)/scale, reduced over i via
// v_wmma_f32_16x16x32_f16 (f16 operands, f32 accumulate). Per-o parameters are
// staged once per block into LDS as (1/scale, -bias/scale, weight), so the
// per-element chain is 1 v_fma + 3 v_mul + v_cos + v_exp + 1 v_mul (+ packed
// f16 cvt). Transcendental args are pre-folded into single multiplies
// (5/2pi for v_cos's normalized input, -0.5*log2e for v_exp's exp2 input).
// Working set (~3 MB) is L2-resident (192 MB), so the kernel is purely
// VALU/transcendental bound; the WMMA pipe absorbs the 128-wide reductions.

typedef __attribute__((ext_vector_type(16))) _Float16 v16h;
typedef __attribute__((ext_vector_type(8)))  float    v8f;

#define OMEGA0 5.0f
// OMEGA0 / (2*pi): v_cos_f32 takes input in units of 2*pi radians.
#define COS_PRESCALE 0.79577471545947667884f
// -0.5 * log2(e): exp(-u^2/2) == exp2(u * (u * EXP2_PRESCALE))
#define EXP2_PRESCALE -0.72134752044448170368f

// One wave handles (o, b_tile of 16 rows). Block = 4 waves = 4 b-tiles, one o.
__global__ __launch_bounds__(128)
void wkan_psi_wmma_kernel(const float* __restrict__ x,
                          const float* __restrict__ scale,
                          const float* __restrict__ bias,
                          const float* __restrict__ weight,
                          float* __restrict__ y,   // [B][O] pre-BN activations
                          int I, int O)
{
    extern __shared__ float smem[];
    float* sR   = smem;          // 1/scale[o][i]
    float* sNbr = smem + I;      // -bias[o][i]/scale[o][i]
    float* sW   = smem + 2 * I;  // weight[o][i]

    const int o = blockIdx.x;    // output feature (uniform per block)

    // Cooperative stage of per-o parameters (I floats each).
    for (int i = threadIdx.x; i < I; i += blockDim.x) {
        const size_t oi = (size_t)o * I + i;
        const float  r  = __fdividef(1.0f, scale[oi]);
        sR[i]   = r;
        sNbr[i] = -bias[oi] * r;
        sW[i]   = weight[oi];
    }
    __syncthreads();

    const int lane = threadIdx.x & 31;
    const int wave = threadIdx.x >> 5;
    const int b0   = (blockIdx.y * 4 + wave) * 16;     // batch tile base
    const int h    = lane >> 4;                        // lane half
    const int m    = lane & 15;                        // A-row within tile

    const float* __restrict__ xrow = x + (size_t)(b0 + m) * I;

    v8f acc = {0.f, 0.f, 0.f, 0.f, 0.f, 0.f, 0.f, 0.f};

    const int nchunks = I >> 5;                        // K=32 per WMMA
    #pragma unroll 4
    for (int c = 0; c < nchunks; ++c) {
        const int ibase = c << 5;

        v16h a;   // A fragment: psi values, ISA 16-bit A 16x32 layout
        v16h bm;  // B fragment: w_o replicated across all 16 columns

        #pragma unroll
        for (int s = 0; s < 16; ++s) {
            // A-matrix K index for (slot s, lane half h)
            const int kA = (s < 8) ? (s + 8 * h) : (s + 8 + 8 * h);
            const int iA = ibase + kA;
            const float u  = fmaf(xrow[iA], sR[iA], sNbr[iA]);
            const float cn = __builtin_amdgcn_cosf(u * COS_PRESCALE);
            const float ex = __builtin_amdgcn_exp2f(u * (u * EXP2_PRESCALE));
            a[s] = (_Float16)(cn * ex);
            // B-matrix K index for (slot s, lane half h)
            const int kB = (s < 8) ? (s + 8 * h) : (s + 8 + 8 * h);
            bm[s] = (_Float16)sW[ibase + kB];
        }

        // (neg_a, A, neg_b, B, c_mod, C, reuse_a, reuse_b)
        acc = __builtin_amdgcn_wmma_f32_16x16x32_f16(
                  false, a, false, bm, (short)0, acc, false, false);
    }

    // All 16 D-columns are identical; lanes 0 (rows 0..7) and 16 (rows 8..15)
    // emit the y column for this (b_tile, o).
    if (m == 0) {
        float* yp = y + (size_t)(b0 + 8 * h) * O + o;
        yp[0 * (size_t)O] = acc[0];
        yp[1 * (size_t)O] = acc[1];
        yp[2 * (size_t)O] = acc[2];
        yp[3 * (size_t)O] = acc[3];
        yp[4 * (size_t)O] = acc[4];
        yp[5 * (size_t)O] = acc[5];
        yp[6 * (size_t)O] = acc[6];
        yp[7 * (size_t)O] = acc[7];
    }
}

// Per-output-column batch statistics; folds gamma/beta into scale/shift.
// Block = 1024 threads = 32 o-columns x 32 b-threads; coalesced y reads.
__global__ __launch_bounds__(1024)
void wkan_bn_stats_kernel(const float* __restrict__ y,
                          const float* __restrict__ gamma,
                          const float* __restrict__ beta,
                          float* __restrict__ bnA,   // gamma*rstd        [O]
                          float* __restrict__ bnB,   // beta - mean*bnA   [O]
                          int B, int O)
{
    __shared__ float s_sum[32][33];
    __shared__ float s_sq[32][33];

    const int ot = threadIdx.x & 31;
    const int bt = threadIdx.x >> 5;
    const int o  = blockIdx.x * 32 + ot;

    float sum = 0.f, sq = 0.f;
    for (int b = bt; b < B; b += 32) {
        const float v = y[(size_t)b * O + o];
        sum += v;
        sq  += v * v;
    }
    s_sum[bt][ot] = sum;
    s_sq[bt][ot]  = sq;
    __syncthreads();

    #pragma unroll
    for (int off = 16; off > 0; off >>= 1) {
        if (bt < off) {
            s_sum[bt][ot] += s_sum[bt + off][ot];
            s_sq[bt][ot]  += s_sq[bt + off][ot];
        }
        __syncthreads();
    }

    if (bt == 0) {
        const float invB = 1.0f / (float)B;
        const float mean = s_sum[0][ot] * invB;
        const float var  = s_sq[0][ot] * invB - mean * mean;  // biased var
        const float rstd = rsqrtf(var + 1e-5f);
        const float a    = gamma[o] * rstd;
        bnA[o] = a;
        bnB[o] = beta[o] - mean * a;
    }
}

// In-place BN apply: out = y*bnA[o] + bnB[o], float4 vectorized (O % 4 == 0).
__global__ __launch_bounds__(256)
void wkan_bn_apply_kernel(float* __restrict__ y,
                          const float* __restrict__ bnA,
                          const float* __restrict__ bnB,
                          int O, size_t n4)
{
    const size_t t = (size_t)blockIdx.x * blockDim.x + threadIdx.x;
    if (t >= n4) return;
    const size_t idx = t * 4;
    const int o = (int)(idx % (size_t)O);

    float4 v = *(const float4*)(y + idx);
    v.x = v.x * bnA[o + 0] + bnB[o + 0];
    v.y = v.y * bnA[o + 1] + bnB[o + 1];
    v.z = v.z * bnA[o + 2] + bnB[o + 2];
    v.w = v.w * bnA[o + 3] + bnB[o + 3];
    *(float4*)(y + idx) = v;
}

extern "C" void kernel_launch(void* const* d_in, const int* in_sizes, int n_in,
                              void* d_out, int out_size, void* d_ws, size_t ws_size,
                              hipStream_t stream)
{
    const float* x      = (const float*)d_in[0];  // [B,I]
    const float* scale  = (const float*)d_in[1];  // [O,I]
    const float* bias   = (const float*)d_in[2];  // [O,I]
    const float* weight = (const float*)d_in[3];  // [O,I]
    const float* gamma  = (const float*)d_in[4];  // [O]
    const float* beta   = (const float*)d_in[5];  // [O]

    const int O = in_sizes[4];                 // 512
    const int I = in_sizes[3] / O;             // 128
    const int B = in_sizes[0] / I;             // 4096

    float* y   = (float*)d_out;                // pre-BN activations, then final
    float* bnA = (float*)d_ws;                 // [O]
    float* bnB = bnA + O;                      // [O]

    // Kernel 1: psi + WMMA reduction. grid = (O, B/64), 4 waves/block,
    // dynamic LDS = 3*I floats for staged per-o parameters.
    dim3 g1((unsigned)O, (unsigned)(B / 64));
    const size_t shmem = (size_t)3 * I * sizeof(float);
    wkan_psi_wmma_kernel<<<g1, 128, shmem, stream>>>(x, scale, bias, weight, y, I, O);

    // Kernel 2a: batch stats per output feature.
    wkan_bn_stats_kernel<<<O / 32, 1024, 0, stream>>>(y, gamma, beta, bnA, bnB, B, O);

    // Kernel 2b: in-place normalize.
    const size_t n4 = (size_t)B * O / 4;
    wkan_bn_apply_kernel<<<(unsigned)((n4 + 255) / 256), 256, 0, stream>>>(y, bnA, bnB, O, n4);
}
